// TransformerAttention_9079560863881
// MI455X (gfx1250) — matrix operand order repaired and checked by reading, not compiled, not run on hardware
//
#include <hip/hip_runtime.h>
#include <hip/hip_bf16.h>

typedef _Float16 f16;
typedef __attribute__((ext_vector_type(16))) _Float16 v16h;
typedef __attribute__((ext_vector_type(8)))  _Float16 v8h;
typedef __attribute__((ext_vector_type(8)))  float    v8f;
typedef __attribute__((ext_vector_type(4)))  unsigned int v4u;
typedef __attribute__((ext_vector_type(8)))  int      v8i;
typedef __attribute__((ext_vector_type(4)))  int      v4i;

#define NSEQ   2048
#define BATCH  2
#define NH     16
#define KVH    4
#define DHEAD  64
#define DMODEL 1024
#define MROWS  (BATCH*NSEQ)   /* 4096 token rows */
#define KVDIM  (KVH*DHEAD)    /* 256 */

#if defined(__has_builtin)
#if __has_builtin(__builtin_amdgcn_tensor_load_to_lds)
#define HAVE_TDM 1
#endif
#endif
#if defined(HAVE_TDM) && (__clang_major__ >= 23)
#define TDM_6ARG 1
#endif

// ---------------------------------------------------------------------------
// WMMA helpers (CDNA5 wave32, v_wmma_f32_16x16x32_f16)
// ---------------------------------------------------------------------------
__device__ __forceinline__ v8f wmma_f16f32(v16h a, v16h b, v8f c) {
  return __builtin_amdgcn_wmma_f32_16x16x32_f16(false, a, false, b, (short)0, c,
                                                false, false);
}

// A-fragment: 16(M) x 32(K) f16 from row-major [m][k] storage.
// lanes 0-15 hold M=lane; halves 0..7 -> K=kh*8+0..7, halves 8..15 ->
// K=16+kh*8+0..7 (kh=lane>>4). Two contiguous 16-byte runs per lane.
__device__ __forceinline__ v16h load_A(const f16* base, int ld) {
  const int lane = threadIdx.x & 31;
  const int m = lane & 15;
  const int kh = lane >> 4;
  const f16* p = base + (size_t)m * ld + kh * 8;
  v8h lo = *(const v8h*)(p);
  v8h hi = *(const v8h*)(p + 16);
  return __builtin_shufflevector(lo, hi, 0,1,2,3,4,5,6,7,8,9,10,11,12,13,14,15);
}

// B-fragment: 32(K) x 16(N) f16 from N-major storage mem[n*ld + k].
// lanes 0-15 -> N=lane; halves 0..7 -> K=kh*8+0..7, halves 8..15 -> K=16+kh*8+0..7.
__device__ __forceinline__ v16h load_B(const f16* base, int ld) {
  const int lane = threadIdx.x & 31;
  const int n = lane & 15;
  const int kh = lane >> 4;
  const f16* p = base + (size_t)n * ld + kh * 8;
  v8h lo = *(const v8h*)(p);
  v8h hi = *(const v8h*)(p + 16);
  return __builtin_shufflevector(lo, hi, 0,1,2,3,4,5,6,7,8,9,10,11,12,13,14,15);
}

// RoPE factor cs = cos(ang) + sign(pos)*sin(ang); sign = -1 on even positions.
__device__ __forceinline__ float rope_cs(int pos, int ai) {
  float theta = __expf(-0.28782313662425572f * (float)(ai >> 1));
  float ang = (float)pos * theta;
  float s, c;
  __sincosf(ang, &s, &c);
  return c + ((pos & 1) ? s : -s);
}

#ifdef HAVE_TDM
// ---------------------------------------------------------------------------
// TDM: DMA one 128-row x 32-half f16 tile (row stride 1024 halfs in memory)
// into LDS with 8-half padding per row (LDS row stride = 40 halfs).
// D# per CDNA5 ISA ch.8:
//  group0: count=1 | lds_addr | global_addr(57b, bytes) | type=2
//  group1: data_size=1(2B), pad_enable, pad_interval=3 (16 DW = one 64B row),
//          pad_amount=3 (4 DW = 16B), tensor_dim0=1024, tensor_dim1=128,
//          tile_dim0=32, tile_dim1=128, tensor_dim0_stride=1024
// ---------------------------------------------------------------------------
__device__ __forceinline__ void tdm_load_tile(unsigned lds_off, const f16* gptr) {
  unsigned long long ga = (unsigned long long)(uintptr_t)gptr;
  v4u g0;
  g0[0] = 1u;                                               // count=1, no gather
  g0[1] = lds_off;                                          // lds_addr (bytes)
  g0[2] = (unsigned)ga;                                     // global_addr[31:0]
  g0[3] = (unsigned)((ga >> 32) & 0x01FFFFFFull) | (2u << 30); // addr[56:32]|type=2
  v8i g1;
  g1[0] = (1 << 16) | (1 << 20) | (3 << 22) | (3 << 25);    // 2B, pad 16DW/4DW
  g1[1] = (int)(1024u << 16);                               // tensor_dim0 = 1024
  g1[2] = (int)(128u << 16);                                // tensor_dim1 = 128
  g1[3] = (int)(32u << 16);                                 // tile_dim0 = 32
  g1[4] = 128;                                              // tile_dim1 = 128
  g1[5] = 1024;                                             // dim0_stride = 1024
  g1[6] = 0;
  g1[7] = 0;
  v4i z4 = {};
#ifdef TDM_6ARG
  v8i z8 = {};
  __builtin_amdgcn_tensor_load_to_lds(g0, g1, z4, z4, z8, 0);
#else
  __builtin_amdgcn_tensor_load_to_lds(g0, g1, z4, z4, 0);
#endif
}
#endif

// ---------------------------------------------------------------------------
// RMSNorm: one row (1024 f32) per 256-thread block -> f16 x_norm
// ---------------------------------------------------------------------------
__global__ __launch_bounds__(256)
void rmsnorm_kernel(const float* __restrict__ x, const float* __restrict__ w,
                    f16* __restrict__ out) {
  __shared__ float red[256];
  const int row = blockIdx.x;
  const float* xr = x + (size_t)row * DMODEL;
  float ss = 0.f;
  for (int i = threadIdx.x; i < DMODEL; i += 256) { float v = xr[i]; ss += v * v; }
  red[threadIdx.x] = ss;
  __syncthreads();
  for (int s = 128; s > 0; s >>= 1) {
    if (threadIdx.x < s) red[threadIdx.x] += red[threadIdx.x + s];
    __syncthreads();
  }
  const float scale = rsqrtf(red[0] * (1.0f / DMODEL) + 1.1920929e-7f);
  for (int i = threadIdx.x; i < DMODEL; i += 256)
    out[(size_t)row * DMODEL + i] = (f16)(xr[i] * scale * w[i]);
}

__global__ __launch_bounds__(256)
void f32_to_f16_kernel(const float* __restrict__ in, f16* __restrict__ out, int count) {
  int i = blockIdx.x * 256 + threadIdx.x;
  if (i < count) out[i] = (f16)in[i];
}

// ---------------------------------------------------------------------------
// WMMA GEMM: C[M,N] = A[M,K] * B[N,K]^T, M=4096, K=1024, N from grid.
// Block tile 128x128, 8 waves, wave tile 32x64 (2x4 of 16x16).
// Tile staging: TDM double-buffered pipeline (wave 0 issues DMA for tile i+1
// while all waves compute tile i; s_wait_tensorcnt<=2 retires tile i).
// MODE 0: Q -> f16 q16[b][h][n][a] (RoPE)
// MODE 1: K -> f32 kv_cache (pre-RoPE) + f16 k16[b][kh][n][a] (RoPE)
// MODE 2: V -> f32 kv_cache + f16 v16[b][kh][n][a]
// MODE 3: O -> f32 d_out = resid + acc
// ---------------------------------------------------------------------------
template <int MODE>
__global__ __launch_bounds__(256)
void gemm_wmma_kernel(const f16* __restrict__ A, const f16* __restrict__ B,
                      f16* __restrict__ hout, float* __restrict__ fout,
                      const float* __restrict__ resid) {
#ifdef HAVE_TDM
  __shared__ f16 Alds[2][128 * 40];
  __shared__ f16 Blds[2][128 * 40];
#else
  __shared__ f16 Alds[1][128 * 40];
  __shared__ f16 Blds[1][128 * 40];
#endif
  const int tid = threadIdx.x;
  const int lane = tid & 31, wid = tid >> 5;
  const int m0 = (wid & 3) * 32;
  const int n0 = (wid >> 2) * 64;
  const int rowA0 = blockIdx.y * 128;
  const int rowB0 = blockIdx.x * 128;

  v8f acc[2][4];
  const v8f vzero = {};
#pragma unroll
  for (int i = 0; i < 2; ++i)
#pragma unroll
    for (int j = 0; j < 4; ++j) acc[i][j] = vzero;

#ifdef HAVE_TDM
  const f16* Abase = A + (size_t)rowA0 * DMODEL;
  const f16* Bbase = B + (size_t)rowB0 * DMODEL;
  if (tid == 0) {                          // prologue: DMA tile 0 into buffer 0
    tdm_load_tile((unsigned)(size_t)&Alds[0][0], Abase);
    tdm_load_tile((unsigned)(size_t)&Blds[0][0], Bbase);
  }
#endif

  for (int it = 0; it < DMODEL / 32; ++it) {
    const int kt = it * 32;
#ifdef HAVE_TDM
    const int buf = it & 1;
    if (tid == 0) {
      if (it + 1 < DMODEL / 32) {          // DMA next tile into other buffer
        tdm_load_tile((unsigned)(size_t)&Alds[buf ^ 1][0], Abase + kt + 32);
        tdm_load_tile((unsigned)(size_t)&Blds[buf ^ 1][0], Bbase + kt + 32);
        __builtin_amdgcn_s_wait_tensorcnt(2);   // current tile's 2 DMAs done
      } else {
        __builtin_amdgcn_s_wait_tensorcnt(0);
      }
    }
    __syncthreads();                        // tile `it` visible to all waves
#else
    const int buf = 0;
    __syncthreads();
    for (int i = tid; i < 512; i += 256) {  // 128 rows x 4 vec8 cols
      int r = i >> 2, c = (i & 3) << 3;
      *(v8h*)&Alds[0][r * 40 + c] = *(const v8h*)&A[(size_t)(rowA0 + r) * DMODEL + kt + c];
      *(v8h*)&Blds[0][r * 40 + c] = *(const v8h*)&B[(size_t)(rowB0 + r) * DMODEL + kt + c];
    }
    if (kt + 32 < DMODEL) {
      __builtin_prefetch(&A[(size_t)(rowA0 + (tid >> 1)) * DMODEL + kt + 32], 0, 0);
      __builtin_prefetch(&B[(size_t)(rowB0 + (tid >> 1)) * DMODEL + kt + 32], 0, 0);
    }
    __syncthreads();
#endif

    v16h a0 = load_A(&Alds[buf][(size_t)m0 * 40], 40);
    v16h a1 = load_A(&Alds[buf][(size_t)(m0 + 16) * 40], 40);
#pragma unroll
    for (int j = 0; j < 4; ++j) {
      v16h bf = load_B(&Blds[buf][(size_t)(n0 + j * 16) * 40], 40);
      acc[0][j] = wmma_f16f32(a0, bf, acc[0][j]);
      acc[1][j] = wmma_f16f32(a1, bf, acc[1][j]);
    }
#ifdef HAVE_TDM
    __syncthreads();   // all waves done with buffer before it is DMA-overwritten
#endif
  }

  // Epilogue. C layout: VGPR r holds M = r + (lane>>4)*8, N = lane&15.
  const int halfsel = lane >> 4, lncol = lane & 15;
#pragma unroll
  for (int i = 0; i < 2; ++i)
#pragma unroll
    for (int j = 0; j < 4; ++j)
#pragma unroll
      for (int r = 0; r < 8; ++r) {
        const int gm = rowA0 + m0 + i * 16 + r + halfsel * 8;
        const int gn = rowB0 + n0 + j * 16 + lncol;
        const float v = acc[i][j][r];
        if (MODE == 0) {
          const int bi = gm >> 11, pos = gm & (NSEQ - 1);
          const int h = gn >> 6, ai = gn & 63;
          hout[(((size_t)bi * NH + h) * NSEQ + pos) * DHEAD + ai] =
              (f16)(v * rope_cs(pos, ai));
        } else if (MODE == 1) {
          const int bi = gm >> 11, pos = gm & (NSEQ - 1);
          fout[((size_t)(0 * BATCH + bi) * NSEQ + pos) * KVDIM + gn] = v;
          const int kh = gn >> 6, ai = gn & 63;
          hout[(((size_t)bi * KVH + kh) * NSEQ + pos) * DHEAD + ai] =
              (f16)(v * rope_cs(pos, ai));
        } else if (MODE == 2) {
          const int bi = gm >> 11, pos = gm & (NSEQ - 1);
          fout[((size_t)(1 * BATCH + bi) * NSEQ + pos) * KVDIM + gn] = v;
          const int kh = gn >> 6, ai = gn & 63;
          hout[(((size_t)bi * KVH + kh) * NSEQ + pos) * DHEAD + ai] = (f16)v;
        } else {
          fout[(size_t)gm * DMODEL + gn] = resid[(size_t)gm * DMODEL + gn] + v;
        }
      }
}

// ---------------------------------------------------------------------------
// Flash attention: grid (n/128 query tiles, b*h). 8 waves; wave w owns
// queries qt*128 + [w*16, w*16+16). 64-key K/V tiles in LDS (V transposed),
// online softmax with shfl-width-16 row reductions, P restaged as f16.
// ---------------------------------------------------------------------------
__global__ __launch_bounds__(256)
void flash_attn_kernel(const f16* __restrict__ q16, const f16* __restrict__ k16,
                       const f16* __restrict__ v16, f16* __restrict__ s16) {
  __shared__ f16 Klds[64 * 72];     // [key][a]
  __shared__ f16 Vlds[64 * 72];     // transposed: [a][key]
  __shared__ f16 Plds[128 * 72];    // [query][key] f16 probabilities
  const int tid = threadIdx.x, lane = tid & 31, wid = tid >> 5;
  const int qt = blockIdx.x;
  const int bh = (int)blockIdx.y;
  const int bi = bh >> 4, h = bh & 15;
  const int kh = h & (KVH - 1);     // reference uses jnp.tile -> h % KV_HEADS
  const int halfsel = lane >> 4, lncol = lane & 15;

  const f16* qbase = q16 + ((((size_t)bi * NH + h) * NSEQ) + qt * 128 + wid * 16) * DHEAD;
  const f16* kbase = k16 + (((size_t)bi * KVH + kh) * NSEQ) * DHEAD;
  const f16* vbase = v16 + (((size_t)bi * KVH + kh) * NSEQ) * DHEAD;

  v16h qa0 = load_A(qbase, DHEAD);
  v16h qa1 = load_A(qbase + 32, DHEAD);

  v8f oacc[4];
  const v8f vzero = {};
#pragma unroll
  for (int j = 0; j < 4; ++j) oacc[j] = vzero;
  float mrow[8], lrow[8];
#pragma unroll
  for (int r = 0; r < 8; ++r) { mrow[r] = -1e38f; lrow[r] = 0.f; }

  const float scale = 0.125f;       // 1/sqrt(64)
  const int ktmax = 2 * qt + 1;

  for (int kt = 0; kt <= ktmax; ++kt) {
    __syncthreads();
    for (int i = tid; i < 512; i += 256) {           // 64 keys x 8 vec8 cols
      int key = i >> 3, c = (i & 7) << 3;
      *(v8h*)&Klds[key * 72 + c] =
          *(const v8h*)&kbase[(size_t)(kt * 64 + key) * DHEAD + c];
      v8h vv = *(const v8h*)&vbase[(size_t)(kt * 64 + key) * DHEAD + c];
#pragma unroll
      for (int e = 0; e < 8; ++e) Vlds[(c + e) * 72 + key] = vv[e];
    }
    __syncthreads();

    v8f sc[4];
#pragma unroll
    for (int nt = 0; nt < 4; ++nt) sc[nt] = vzero;
#pragma unroll
    for (int nt = 0; nt < 4; ++nt) {
      v16h b0 = load_B(&Klds[(size_t)(nt * 16) * 72], 72);
      v16h b1 = load_B(&Klds[(size_t)(nt * 16) * 72 + 32], 72);
      sc[nt] = wmma_f16f32(qa0, b0, sc[nt]);
      sc[nt] = wmma_f16f32(qa1, b1, sc[nt]);
    }

#pragma unroll
    for (int nt = 0; nt < 4; ++nt)
#pragma unroll
      for (int r = 0; r < 8; ++r) {
        const int key = kt * 64 + nt * 16 + lncol;
        const int qidx = qt * 128 + wid * 16 + r + halfsel * 8;
        sc[nt][r] = (key <= qidx) ? sc[nt][r] * scale : -1e38f;
      }

#pragma unroll
    for (int r = 0; r < 8; ++r) {
      float mx = fmaxf(fmaxf(sc[0][r], sc[1][r]), fmaxf(sc[2][r], sc[3][r]));
      mx = fmaxf(mx, __shfl_xor(mx, 1, 16));
      mx = fmaxf(mx, __shfl_xor(mx, 2, 16));
      mx = fmaxf(mx, __shfl_xor(mx, 4, 16));
      mx = fmaxf(mx, __shfl_xor(mx, 8, 16));
      const float mnew = fmaxf(mrow[r], mx);
      const float alpha = __expf(mrow[r] - mnew);
      mrow[r] = mnew;
      float rs = 0.f;
#pragma unroll
      for (int nt = 0; nt < 4; ++nt) {
        float p = __expf(sc[nt][r] - mnew);
        sc[nt][r] = p;
        rs += p;
      }
      rs += __shfl_xor(rs, 1, 16);
      rs += __shfl_xor(rs, 2, 16);
      rs += __shfl_xor(rs, 4, 16);
      rs += __shfl_xor(rs, 8, 16);
      lrow[r] = lrow[r] * alpha + rs;
#pragma unroll
      for (int j = 0; j < 4; ++j) oacc[j][r] *= alpha;
    }

#pragma unroll
    for (int nt = 0; nt < 4; ++nt)
#pragma unroll
      for (int r = 0; r < 8; ++r)
        Plds[(wid * 16 + r + halfsel * 8) * 72 + nt * 16 + lncol] = (f16)sc[nt][r];
    __syncthreads();

    v16h pa0 = load_A(&Plds[(size_t)(wid * 16) * 72], 72);
    v16h pa1 = load_A(&Plds[(size_t)(wid * 16) * 72 + 32], 72);
#pragma unroll
    for (int j = 0; j < 4; ++j) {
      v16h b0 = load_B(&Vlds[(size_t)(j * 16) * 72], 72);
      v16h b1 = load_B(&Vlds[(size_t)(j * 16) * 72 + 32], 72);
      oacc[j] = wmma_f16f32(pa0, b0, oacc[j]);
      oacc[j] = wmma_f16f32(pa1, b1, oacc[j]);
    }
  }

#pragma unroll
  for (int j = 0; j < 4; ++j)
#pragma unroll
    for (int r = 0; r < 8; ++r) {
      const int qidx = qt * 128 + wid * 16 + r + halfsel * 8;
      const float o = oacc[j][r] / lrow[r];
      s16[(((size_t)bi * NSEQ + qidx) * NH + h) * DHEAD + j * 16 + lncol] = (f16)o;
    }
}

// ---------------------------------------------------------------------------
// Host launch
// ---------------------------------------------------------------------------
extern "C" void kernel_launch(void* const* d_in, const int* in_sizes, int n_in,
                              void* d_out, int out_size, void* d_ws, size_t ws_size,
                              hipStream_t stream) {
  const float* x      = (const float*)d_in[0];
  const float* w_q    = (const float*)d_in[1];
  const float* w_k    = (const float*)d_in[2];
  const float* w_v    = (const float*)d_in[3];
  const float* w_o    = (const float*)d_in[4];
  const float* norm_w = (const float*)d_in[5];
  float* out = (float*)d_out;

  f16* ws = (f16*)d_ws;
  size_t off = 0;
  f16* xn   = ws + off; off += (size_t)MROWS * DMODEL;
  f16* wq16 = ws + off; off += (size_t)DMODEL * DMODEL;
  f16* wk16 = ws + off; off += (size_t)KVDIM * DMODEL;
  f16* wv16 = ws + off; off += (size_t)KVDIM * DMODEL;
  f16* wo16 = ws + off; off += (size_t)DMODEL * DMODEL;
  f16* q16  = ws + off; off += (size_t)BATCH * NH  * NSEQ * DHEAD;
  f16* k16  = ws + off; off += (size_t)BATCH * KVH * NSEQ * DHEAD;
  f16* v16  = ws + off; off += (size_t)BATCH * KVH * NSEQ * DHEAD;
  f16* s16  = ws + off; off += (size_t)MROWS * DMODEL;

  rmsnorm_kernel<<<MROWS, 256, 0, stream>>>(x, norm_w, xn);
  f32_to_f16_kernel<<<(DMODEL * DMODEL) / 256, 256, 0, stream>>>(w_q, wq16, DMODEL * DMODEL);
  f32_to_f16_kernel<<<(KVDIM  * DMODEL) / 256, 256, 0, stream>>>(w_k, wk16, KVDIM * DMODEL);
  f32_to_f16_kernel<<<(KVDIM  * DMODEL) / 256, 256, 0, stream>>>(w_v, wv16, KVDIM * DMODEL);
  f32_to_f16_kernel<<<(DMODEL * DMODEL) / 256, 256, 0, stream>>>(w_o, wo16, DMODEL * DMODEL);

  float* kvout = out + (size_t)MROWS * DMODEL;   // kv_cache region of d_out
  dim3 blk(256);
  gemm_wmma_kernel<0><<<dim3(DMODEL / 128, MROWS / 128), blk, 0, stream>>>(xn, wq16, q16, nullptr, nullptr);
  gemm_wmma_kernel<1><<<dim3(KVDIM  / 128, MROWS / 128), blk, 0, stream>>>(xn, wk16, k16, kvout, nullptr);
  gemm_wmma_kernel<2><<<dim3(KVDIM  / 128, MROWS / 128), blk, 0, stream>>>(xn, wv16, v16, kvout, nullptr);
  flash_attn_kernel<<<dim3(NSEQ / 128, BATCH * NH), blk, 0, stream>>>(q16, k16, v16, s16);
  gemm_wmma_kernel<3><<<dim3(DMODEL / 128, MROWS / 128), blk, 0, stream>>>(s16, wo16, nullptr, out, x);
}
